// SwinAttentionLayer_46677704572977
// MI455X (gfx1250) — compile-verified, matrix operand order checked by reading, not run on hardware
//
#include <hip/hip_runtime.h>
#include <hip/hip_bf16.h>

typedef __attribute__((ext_vector_type(16))) __bf16 v16bf;
typedef __attribute__((ext_vector_type(8)))  float  v8f;
typedef __attribute__((ext_vector_type(4)))  float  v4f;

#define CCH   96          // channels
#define L49   49          // tokens per window
#define LP    64          // padded tokens per window
#define HW    224
#define NTOK  401408      // 8*224*224
#define NWIN  8192        // 8 * 32 * 32

// ---------------------------------------------------------------------------
// WMMA helpers (CDNA5 16x16x32 bf16, f32 accumulate)
// ---------------------------------------------------------------------------
__device__ __forceinline__ v8f wmma_bf16(v16bf a, v16bf b, v8f c) {
  // (neg_a, A, neg_b, B, c_mod, C, reuse_a, reuse_b)
  return __builtin_amdgcn_wmma_f32_16x16x32_bf16(false, a, false, b, (short)0, c,
                                                 false, false);
}

// Load a 16x32 bf16 fragment (A layout; also used for B from row-major [N,K]).
// ISA 7.12.2: lanes 0-15 hold row M=lane with K={0..7,16..23} in VGPR halves;
// lanes 16-31 hold row M=lane-16 with K={8..15,24..31}.
__device__ __forceinline__ v16bf load_frag(const __bf16* base, int ld,
                                           int row_base, int col_base) {
  const int lane = threadIdx.x & 31;
  const int r    = lane & 15;
  const int kb   = (lane >> 4) << 3;   // 0 or 8
  const __bf16* p = base + (size_t)(row_base + r) * ld + col_base;
  v16bf f;
#pragma unroll
  for (int j = 0; j < 4; ++j) {
    f[2*j]       = p[kb + 2*j];
    f[2*j + 1]   = p[kb + 2*j + 1];
    f[8 + 2*j]   = p[16 + kb + 2*j];
    f[8 + 2*j+1] = p[16 + kb + 2*j + 1];
  }
  return f;
}

// ---------------------------------------------------------------------------
// Kernel 0a: fp32 -> bf16 weight conversion (tiny, L2-resident afterwards)
// ---------------------------------------------------------------------------
__global__ void swin_wconv_kernel(const float* __restrict__ qkvw,
                                  const float* __restrict__ projw,
                                  const float* __restrict__ m1,
                                  const float* __restrict__ m2,
                                  __bf16* bq, __bf16* bp, __bf16* b1, __bf16* b2) {
  int i = blockIdx.x * blockDim.x + threadIdx.x;
  if (i < 288 * 96) bq[i] = (__bf16)qkvw[i];
  if (i < 96 * 96)  bp[i] = (__bf16)projw[i];
  if (i < 384 * 96) { b1[i] = (__bf16)m1[i]; b2[i] = (__bf16)m2[i]; }
}

// ---------------------------------------------------------------------------
// Kernel 0b: NCHW -> NHWC residual base + LN1 + cyclic shift + window partition
// hbuf: [NTOK, 96] f32 (NHWC residual stream)
// xn:   [NWIN, 49, 96] bf16 (normalized, shifted, window-partitioned)
// ---------------------------------------------------------------------------
__global__ __launch_bounds__(256)
void swin_ln1_kernel(const float* __restrict__ x,
                     const float* __restrict__ w,
                     const float* __restrict__ b,
                     float* __restrict__ hbuf,
                     __bf16* __restrict__ xn) {
  int t = blockIdx.x * blockDim.x + threadIdx.x;
  if (t >= NTOK) return;
  int bb  = t / (HW * HW);
  int rem = t - bb * (HW * HW);
  int hh  = rem / HW;
  int ww  = rem - hh * HW;
  const float* px = x + (size_t)bb * CCH * HW * HW + rem;   // stride HW*HW per ch

  float mu = 0.f;
  for (int c = 0; c < CCH; ++c) mu += px[(size_t)c * HW * HW];
  mu *= (1.f / CCH);
  float var = 0.f;
  for (int c = 0; c < CCH; ++c) {
    float d = px[(size_t)c * HW * HW] - mu;
    var += d * d;
  }
  var *= (1.f / CCH);
  float inv = rsqrtf(var + 1e-5f);

  // cyclic shift by (-3,-3): rolled[hr] = x[(hr+3)%224]  =>  hr = (hh+221)%224
  int hr = hh + 221; if (hr >= HW) hr -= HW;
  int wr = ww + 221; if (wr >= HW) wr -= HW;
  int wy = hr / 7, ry = hr - wy * 7;
  int wx = wr / 7, rx = wr - wx * 7;
  int win = (bb * 32 + wy) * 32 + wx;
  int pos = ry * 7 + rx;
  float*  ph = hbuf + (size_t)t * CCH;
  __bf16* pn = xn + ((size_t)win * L49 + pos) * CCH;
  for (int c = 0; c < CCH; ++c) {
    float v = px[(size_t)c * HW * HW];
    ph[c] = v;
    pn[c] = (__bf16)((v - mu) * inv * w[c] + b[c]);
  }
}

// ---------------------------------------------------------------------------
// Kernel 1: fused shifted-window attention. One block per window (8192 blocks).
// async-LDS stage -> QKV GEMM -> QK^T (+bias +mask) -> softmax -> PV -> proj
// ---------------------------------------------------------------------------
__global__ __launch_bounds__(256)
void swin_attn_kernel(const __bf16* __restrict__ xn,
                      const __bf16* __restrict__ wqkv,   // [288,96] bf16
                      const float*  __restrict__ bqkv,   // [288]
                      const float*  __restrict__ rel,    // [169]
                      const __bf16* __restrict__ wproj,  // [96,96] bf16
                      const float*  __restrict__ bproj,  // [96]
                      float* __restrict__ hbuf) {
  __shared__ __align__(16) char smem[64 * 1024];
  __bf16* sX  = (__bf16*)(smem);               // 12 KB  (window tile)
  __bf16* sQ  = (__bf16*)(smem + 12 * 1024);   // 12 KB
  __bf16* sK  = (__bf16*)(smem + 24 * 1024);   // 12 KB
  __bf16* sVT = (__bf16*)(smem + 36 * 1024);   // 12 KB  (V transposed [ch][tok])
  float*  sS  = (float*) (smem + 48 * 1024);   // 16 KB  (scores)
  __bf16* sP  = (__bf16*)(smem);               // aliases sX (dead after QKV)
  __bf16* sO  = (__bf16*)(smem + 12 * 1024);   // aliases sQ (dead after scores)

  const int win  = blockIdx.x;
  const int bb   = win >> 10;
  const int wy   = (win >> 5) & 31;
  const int wx   = win & 31;
  const int tid  = threadIdx.x;
  const int wave = tid >> 5;
  const int lane = tid & 31;
  const int cIdx  = lane & 15;          // C/D fragment: N = lane%16
  const int rBase = (lane >> 4) << 3;   // C/D fragment: M = rBase + vgpr
  const bool edge = (wy == 31) || (wx == 31);   // only edge windows are masked

  // ---- stage window tokens: async DMA global->LDS (ASYNCcnt path) ----
  // 49*96 bf16 = 9408 B = 588 x b128 chunks, VGPR-bypassing.
  {
    const unsigned long long gbase =
        (unsigned long long)(xn + (size_t)win * L49 * CCH);
    for (int i = tid; i < 588; i += 256) {
      unsigned lds_off = (unsigned)(i * 16);
      unsigned long long ga = gbase + (unsigned long long)(i * 16);
      asm volatile("global_load_async_to_lds_b128 %0, %1, off"
                   :: "v"(lds_off), "v"(ga) : "memory");
    }
    // zero-pad rows 49..63 (2880 B = 180 x 16B) with plain LDS stores
    for (int i = tid; i < 180; i += 256)
      *(v4f*)(smem + 9408 + i * 16) = (v4f){0.f, 0.f, 0.f, 0.f};
    asm volatile("s_wait_asynccnt 0x0" ::: "memory");
  }
  __syncthreads();

  // ---- QKV: [64,288] = sX[64,96] @ Wqkv^T ; 72 tiles over 8 waves ----
  // Tile index t is wave-uniform, so the Q/K/V routing below is scalar
  // (each 16-wide N-tile lies entirely inside Q, K, or V since 96 = 6*16).
  for (int t = wave; t < 72; t += 8) {
    int m = t / 18, n = t % 18;
    v8f acc = {};
#pragma unroll
    for (int kk = 0; kk < CCH; kk += 32) {
      v16bf a  = load_frag(sX, CCH, m * 16, kk);
      v16bf bw = load_frag(wqkv, CCH, n * 16, kk);
      acc = wmma_bf16(a, bw, acc);
    }
    const int colB = n * 16;
    const int col  = colB + cIdx;
    const int row0 = m * 16 + rBase;
    const float bias = bqkv[col];
    if (colB < 96) {              // Q tile (pre-scaled by hd^-0.5)
#pragma unroll
      for (int i = 0; i < 8; ++i)
        sQ[(row0 + i) * CCH + col] =
            (__bf16)((acc[i] + bias) * 0.1020620726159658f);
    } else if (colB < 192) {      // K tile
#pragma unroll
      for (int i = 0; i < 8; ++i)
        sK[(row0 + i) * CCH + (col - 96)] = (__bf16)(acc[i] + bias);
    } else {                      // V tile -> transposed [ch][tok]
#pragma unroll
      for (int i = 0; i < 8; ++i)
        sVT[(col - 192) * LP + (row0 + i)] = (__bf16)(acc[i] + bias);
    }
  }
  __syncthreads();

  // ---- scores: S[64,64] = Q @ K^T (+bias +mask) ; 16 tiles ----
  for (int t = wave; t < 16; t += 8) {
    int m = t >> 2, n = t & 3;
    v8f acc = {};
#pragma unroll
    for (int kk = 0; kk < CCH; kk += 32) {
      v16bf a  = load_frag(sQ, CCH, m * 16, kk);
      v16bf bw = load_frag(sK, CCH, n * 16, kk);   // row = key idx, col = ch
      acc = wmma_bf16(a, bw, acc);
    }
    const int col = n * 16 + cIdx;                 // key index
    const int ky  = col / 7, kx = col - ky * 7;
#pragma unroll
    for (int i = 0; i < 8; ++i) {
      int row = m * 16 + rBase + i;                // query index
      float s = acc[i];
      if (col >= L49) {
        s = -1e30f;                                // padded keys -> zero prob
      } else if (row < L49) {
        int qy = row / 7, qx = row - qy * 7;
        s += rel[(qy - ky + 6) * 13 + (qx - kx + 6)];
        if (edge) {                                // wave-uniform branch
          int gqh = wy * 7 + qy, gqw = wx * 7 + qx;
          int gkh = wy * 7 + ky, gkw = wx * 7 + kx;
          int lq = (gqh < 217 ? 0 : (gqh < 221 ? 1 : 2)) * 3 +
                   (gqw < 217 ? 0 : (gqw < 221 ? 1 : 2));
          int lk = (gkh < 217 ? 0 : (gkh < 221 ? 1 : 2)) * 3 +
                   (gkw < 217 ? 0 : (gkw < 221 ? 1 : 2));
          if (lq != lk) s -= 100.0f;
        }
      }
      sS[row * LP + col] = s;
    }
  }
  __syncthreads();

  // ---- softmax rows; write bf16 probs into sP ----
  if (tid < LP) {
    float* rowp = sS + tid * LP;
    float mx = -1e30f;
    for (int j = 0; j < LP; ++j) mx = fmaxf(mx, rowp[j]);
    float sum = 0.f;
    for (int j = 0; j < LP; ++j) { float e = expf(rowp[j] - mx); rowp[j] = e; sum += e; }
    float inv = 1.0f / sum;
    for (int j = 0; j < LP; ++j) sP[tid * LP + j] = (__bf16)(rowp[j] * inv);
  }
  __syncthreads();

  // ---- out = P[64,64] @ V[64,96] ; 24 tiles ----
  for (int t = wave; t < 24; t += 8) {
    int m = t / 6, n = t % 6;
    v8f acc = {};
#pragma unroll
    for (int kk = 0; kk < LP; kk += 32) {
      v16bf a  = load_frag(sP, LP, m * 16, kk);
      v16bf bw = load_frag(sVT, LP, n * 16, kk);   // row = ch, col = key
      acc = wmma_bf16(a, bw, acc);
    }
    const int col  = n * 16 + cIdx;
    const int row0 = m * 16 + rBase;
#pragma unroll
    for (int i = 0; i < 8; ++i)
      sO[(row0 + i) * CCH + col] = (__bf16)acc[i];
  }
  __syncthreads();

  // ---- proj + window-reverse + un-shift + residual accumulate ----
  for (int t = wave; t < 24; t += 8) {
    int m = t / 6, n = t % 6;
    v8f acc = {};
#pragma unroll
    for (int kk = 0; kk < CCH; kk += 32) {
      v16bf a  = load_frag(sO, CCH, m * 16, kk);
      v16bf bw = load_frag(wproj, CCH, n * 16, kk);
      acc = wmma_bf16(a, bw, acc);
    }
    const int col = n * 16 + cIdx;
    const float bias = bproj[col];
#pragma unroll
    for (int i = 0; i < 8; ++i) {
      int row = m * 16 + rBase + i;
      if (row >= L49) continue;
      float v = acc[i] + bias;
      int ry = row / 7, rx = row - ry * 7;
      int hh = wy * 7 + ry + 3; if (hh >= HW) hh -= HW;
      int ww = wx * 7 + rx + 3; if (ww >= HW) ww -= HW;
      size_t idx = (((size_t)bb * HW + hh) * HW + ww) * CCH + col;
      hbuf[idx] += v;   // unique (token,ch) per window element -> race-free
    }
  }
}

// ---------------------------------------------------------------------------
// Kernel 2: fused LN2 + MLP (96->384 GELU ->96) + residual + NHWC->NCHW out.
// One block per 64 tokens (6272 blocks).
// ---------------------------------------------------------------------------
__global__ __launch_bounds__(256)
void swin_mlp_kernel(const float* __restrict__ hbuf,
                     const float* __restrict__ n2w,
                     const float* __restrict__ n2b,
                     const __bf16* __restrict__ w1,   // [384,96] bf16
                     const float*  __restrict__ b1,   // [384]
                     const __bf16* __restrict__ w2,   // [96,384] bf16
                     const float*  __restrict__ b2,   // [96]
                     float* __restrict__ out) {
  __shared__ __bf16 sXn[LP * CCH];    // 12 KB
  __shared__ __bf16 sH[LP * 384];     // 48 KB
  const int t0   = blockIdx.x * LP;
  const int tid  = threadIdx.x;
  const int wave = tid >> 5;
  const int lane = tid & 31;
  const int cIdx  = lane & 15;
  const int rBase = (lane >> 4) << 3;

  if (tid < LP) {
    const float* ph = hbuf + (size_t)(t0 + tid) * CCH;
    float mu = 0.f;
    for (int c = 0; c < CCH; ++c) mu += ph[c];
    mu *= (1.f / CCH);
    float var = 0.f;
    for (int c = 0; c < CCH; ++c) { float d = ph[c] - mu; var += d * d; }
    var *= (1.f / CCH);
    float inv = rsqrtf(var + 1e-5f);
    for (int c = 0; c < CCH; ++c)
      sXn[tid * CCH + c] = (__bf16)((ph[c] - mu) * inv * n2w[c] + n2b[c]);
  }
  __syncthreads();

  // ---- GEMM1: hidden[64,384] = Xn @ W1^T, + bias, exact GELU ; 96 tiles ----
  for (int t = wave; t < 96; t += 8) {
    int m = t / 24, n = t % 24;
    v8f acc = {};
#pragma unroll
    for (int kk = 0; kk < CCH; kk += 32) {
      v16bf a  = load_frag(sXn, CCH, m * 16, kk);
      v16bf bw = load_frag(w1, CCH, n * 16, kk);
      acc = wmma_bf16(a, bw, acc);
    }
    const int col = n * 16 + cIdx;
    const int row0 = m * 16 + rBase;
    const float bias = b1[col];
#pragma unroll
    for (int i = 0; i < 8; ++i) {
      float v = acc[i] + bias;
      v = 0.5f * v * (1.0f + erff(v * 0.70710678118654752f));   // exact GELU
      sH[(row0 + i) * 384 + col] = (__bf16)v;
    }
  }
  __syncthreads();

  // ---- GEMM2: out[64,96] = H @ W2^T + bias + residual -> NCHW ; 24 tiles ----
  for (int t = wave; t < 24; t += 8) {
    int m = t / 6, n = t % 6;
    v8f acc = {};
#pragma unroll
    for (int kk = 0; kk < 384; kk += 32) {
      v16bf a  = load_frag(sH, 384, m * 16, kk);
      v16bf bw = load_frag(w2, 384, n * 16, kk);
      acc = wmma_bf16(a, bw, acc);
    }
    const int col = n * 16 + cIdx;
    const float bias = b2[col];
#pragma unroll
    for (int i = 0; i < 8; ++i) {
      int row = m * 16 + rBase + i;
      int tok = t0 + row;
      float v = acc[i] + bias + hbuf[(size_t)tok * CCH + col];
      int bb  = tok / (HW * HW);
      int rem = tok - bb * (HW * HW);
      out[((size_t)bb * CCH + col) * (HW * HW) + rem] = v;   // NHWC -> NCHW
    }
  }
}

// ---------------------------------------------------------------------------
// Host launcher
// ---------------------------------------------------------------------------
extern "C" void kernel_launch(void* const* d_in, const int* in_sizes, int n_in,
                              void* d_out, int out_size, void* d_ws, size_t ws_size,
                              hipStream_t stream) {
  (void)in_sizes; (void)n_in; (void)out_size; (void)ws_size;
  const float* x     = (const float*)d_in[0];
  const float* n1w   = (const float*)d_in[1];
  const float* n1b   = (const float*)d_in[2];
  const float* qkvw  = (const float*)d_in[3];
  const float* qkvb  = (const float*)d_in[4];
  const float* rel   = (const float*)d_in[5];
  const float* projw = (const float*)d_in[6];
  const float* projb = (const float*)d_in[7];
  const float* n2w   = (const float*)d_in[8];
  const float* n2b   = (const float*)d_in[9];
  const float* m1w   = (const float*)d_in[10];
  const float* m1b   = (const float*)d_in[11];
  const float* m2w   = (const float*)d_in[12];
  const float* m2b   = (const float*)d_in[13];
  float* out = (float*)d_out;

  char* ws = (char*)d_ws;
  size_t off = 0;
  float*  hbuf = (float*)(ws + off);  off += (size_t)NTOK * CCH * 4;   // 154.1 MB
  __bf16* xn   = (__bf16*)(ws + off); off += (size_t)NTOK * CCH * 2;   //  77.1 MB
  __bf16* bq   = (__bf16*)(ws + off); off += (size_t)288 * 96 * 2; off = (off + 255) & ~(size_t)255;
  __bf16* bp   = (__bf16*)(ws + off); off += (size_t)96 * 96 * 2;  off = (off + 255) & ~(size_t)255;
  __bf16* b1w  = (__bf16*)(ws + off); off += (size_t)384 * 96 * 2; off = (off + 255) & ~(size_t)255;
  __bf16* b2w  = (__bf16*)(ws + off); off += (size_t)96 * 384 * 2; off = (off + 255) & ~(size_t)255;

  swin_wconv_kernel<<<(384 * 96 + 255) / 256, 256, 0, stream>>>(
      qkvw, projw, m1w, m2w, bq, bp, b1w, b2w);

  swin_ln1_kernel<<<(NTOK + 255) / 256, 256, 0, stream>>>(
      x, n1w, n1b, hbuf, xn);

  swin_attn_kernel<<<NWIN, 256, 0, stream>>>(
      xn, bq, qkvb, rel, bp, projb, hbuf);

  swin_mlp_kernel<<<NTOK / LP, 256, 0, stream>>>(
      hbuf, n2w, n2b, b1w, m1b, b2w, m2b, out);
}
